// Transformer_36558761623846
// MI455X (gfx1250) — matrix operand order repaired and checked by reading, not compiled, not run on hardware
//
#include <hip/hip_runtime.h>
#include <hip/hip_bf16.h>

// ---------------------------------------------------------------------------
// MI455X (gfx1250, wave32) implementation.
// All GEMM-shaped compute runs on v_wmma_f32_16x16x32_f16 (fp32 accumulate).
// ---------------------------------------------------------------------------

typedef __attribute__((ext_vector_type(16))) _Float16 v16h;
typedef __attribute__((ext_vector_type(8)))  float    v8f;

union Frag16 { v16h v; float4 f4[2]; };

__device__ __forceinline__ v8f vzero8() {
  v8f a = {0.f,0.f,0.f,0.f,0.f,0.f,0.f,0.f};
  return a;
}

// Model constants
#define BB   16
#define TT   40000
#define P0   9997          // conv0 pooled length
#define C0   8
#define LP   2496          // conv1 pooled length (sequence length)
#define DM   256
#define NH   8
#define DH   32
#define M2   (BB*LP)       // 39936 rows for [B*L, D] GEMMs

// ---------------------------------------------------------------------------
// conv0 + relu + mean-pool(4)  (tiny: K=11, Cin=1, Cout=8) -> h0 [B,9997,8]
// ---------------------------------------------------------------------------
__global__ void __launch_bounds__(256)
k_conv0(const float* __restrict__ X, const float* __restrict__ W0,
        const float* __restrict__ B0, float* __restrict__ H0)
{
  int idx = blockIdx.x * 256 + threadIdx.x;            // over B*P0*C0
  if (idx >= BB * P0 * C0) return;
  int c    = idx & 7;
  int rest = idx >> 3;
  int pp   = rest % P0;
  int b    = rest / P0;
  const float* x = X + (size_t)b * TT + pp * 4;
  float accp = 0.f;
#pragma unroll
  for (int p = 0; p < 4; ++p) {
    float s = B0[c];
#pragma unroll
    for (int k = 0; k < 11; ++k) s += x[p + k] * W0[k * 8 + c];
    accp += fmaxf(s, 0.f);
  }
  H0[idx] = 0.25f * accp;
}

// ---------------------------------------------------------------------------
// Generic WMMA GEMM: C[M,N] = op(A[M,K] @ W[K,N] + bias) (+resid)
// 64x64 tile, 128 threads (4 waves), each wave a 32x32 subtile (2x2 frags).
// Requires K%32==0, N%64==0, lda%4==0. M guarded.
// ---------------------------------------------------------------------------
__global__ void __launch_bounds__(128)
k_gemm64(const float* __restrict__ A, int lda,
         const float* __restrict__ W,
         const float* __restrict__ bias,
         const float* __restrict__ resid,
         float* __restrict__ C, int ldc,
         int M, int N, int K, int relu)
{
  __shared__ __align__(16) _Float16 As[64][40];
  __shared__ __align__(16) _Float16 BsT[64][40];

  const int n0   = blockIdx.x * 64;
  const int m0   = blockIdx.y * 64;
  const int tid  = threadIdx.x;
  const int lane = tid & 31;
  const int w    = tid >> 5;
  const int wm   = (w >> 1) * 32;
  const int wn   = (w & 1) * 32;
  const int l15  = lane & 15;
  const int lhi  = lane >> 4;

  v8f acc[2][2];
  acc[0][0] = vzero8(); acc[0][1] = vzero8();
  acc[1][0] = vzero8(); acc[1][1] = vzero8();

  const int  ar      = tid >> 1;
  const int  akh     = (tid & 1) * 16;
  const bool arow_ok = (m0 + ar) < M;
  const int  bkr     = tid >> 2;
  const int  bnb     = (tid & 3) * 16;

  for (int kk = 0; kk < K; kk += 32) {
    __syncthreads();
    {  // stage A tile (fp32 -> f16)
      const float* ap = A + (size_t)(m0 + ar) * lda + kk + akh;
      if (kk + 32 < K) __builtin_prefetch(ap + 32, 0, 1);   // global_prefetch_b8
#pragma unroll
      for (int i = 0; i < 4; ++i) {
        float4 t4 = arow_ok ? *reinterpret_cast<const float4*>(ap + 4 * i)
                            : make_float4(0.f, 0.f, 0.f, 0.f);
        As[ar][akh + 4*i + 0] = (_Float16)t4.x;
        As[ar][akh + 4*i + 1] = (_Float16)t4.y;
        As[ar][akh + 4*i + 2] = (_Float16)t4.z;
        As[ar][akh + 4*i + 3] = (_Float16)t4.w;
      }
    }
    {  // stage B tile transposed: BsT[n][k]
      const float* bp = W + (size_t)(kk + bkr) * N + n0 + bnb;
#pragma unroll
      for (int i = 0; i < 4; ++i) {
        float4 t4 = *reinterpret_cast<const float4*>(bp + 4 * i);
        BsT[bnb + 4*i + 0][bkr] = (_Float16)t4.x;
        BsT[bnb + 4*i + 1][bkr] = (_Float16)t4.y;
        BsT[bnb + 4*i + 2][bkr] = (_Float16)t4.z;
        BsT[bnb + 4*i + 3][bkr] = (_Float16)t4.w;
      }
    }
    __syncthreads();

    Frag16 a[2], b[2];
#pragma unroll
    for (int mi = 0; mi < 2; ++mi) {  // A 16x32 layout: lanes0-15 K0-7/16-23, lanes16-31 K8-15/24-31
      const _Float16* s = &As[wm + 16*mi + l15][lhi * 8];
      a[mi].f4[0] = *reinterpret_cast<const float4*>(s);
      a[mi].f4[1] = *reinterpret_cast<const float4*>(s + 16);
    }
#pragma unroll
    for (int ni = 0; ni < 2; ++ni) {  // B 32x16 layout: lanes0-15 K0-7/16-23, lanes16-31 K8-15/24-31
      const _Float16* s = &BsT[wn + 16*ni + l15][lhi * 8];
      b[ni].f4[0] = *reinterpret_cast<const float4*>(s);
      b[ni].f4[1] = *reinterpret_cast<const float4*>(s + 16);
    }
#pragma unroll
    for (int mi = 0; mi < 2; ++mi)
#pragma unroll
      for (int ni = 0; ni < 2; ++ni)
        acc[mi][ni] = __builtin_amdgcn_wmma_f32_16x16x32_f16(
            false, a[mi].v, false, b[ni].v, (short)0, acc[mi][ni], false, false);
  }

#pragma unroll
  for (int mi = 0; mi < 2; ++mi) {
#pragma unroll
    for (int ni = 0; ni < 2; ++ni) {
      const int   col = n0 + wn + 16*ni + l15;
      const float bv  = bias ? bias[col] : 0.f;
#pragma unroll
      for (int r = 0; r < 8; ++r) {
        const int row = m0 + wm + 16*mi + 8*lhi + r;  // C/D layout: M=r+8*(lane>>4)
        if (row < M) {
          float v = acc[mi][ni][r] + bv;
          if (relu)  v = fmaxf(v, 0.f);
          if (resid) v += resid[(size_t)row * ldc + col];
          C[(size_t)row * ldc + col] = v;
        }
      }
    }
  }
}

// ---------------------------------------------------------------------------
// conv1 as WMMA GEMM (im2col is just stride-8 over h0) + fused relu+pool(4).
// A row m (pre-pool pos) = h0[b, m*8 .. m*8+87]; K=88 zero-padded to 96.
// grid (4, 156, B), tile 64(M-prepool) x 64(N-chan).
// ---------------------------------------------------------------------------
__global__ void __launch_bounds__(128)
k_conv1(const float* __restrict__ H0, const float* __restrict__ W1,
        const float* __restrict__ B1, float* __restrict__ Hout)
{
  __shared__ __align__(16) _Float16 As[64][40];
  __shared__ __align__(16) _Float16 BsT[64][40];

  const int b    = blockIdx.z;
  const int m0   = blockIdx.y * 64;
  const int n0   = blockIdx.x * 64;
  const int tid  = threadIdx.x;
  const int lane = tid & 31;
  const int w    = tid >> 5;
  const int wm   = (w >> 1) * 32;
  const int wn   = (w & 1) * 32;
  const int l15  = lane & 15;
  const int lhi  = lane >> 4;
  const float* Ab = H0 + (size_t)b * (P0 * C0);

  v8f acc[2][2];
  acc[0][0] = vzero8(); acc[0][1] = vzero8();
  acc[1][0] = vzero8(); acc[1][1] = vzero8();

  const int ar  = tid >> 1;
  const int akh = (tid & 1) * 16;
  const int bkr = tid >> 2;
  const int bnb = (tid & 3) * 16;

  for (int kk = 0; kk < 96; kk += 32) {
    __syncthreads();
#pragma unroll
    for (int j = 0; j < 16; ++j) {        // A stage (scalar, K-guarded)
      int kg = kk + akh + j;
      As[ar][akh + j] = (_Float16)((kg < 88) ? Ab[(size_t)(m0 + ar) * 8 + kg] : 0.f);
    }
    {
      int kg = kk + bkr;                  // B stage transposed, K-guarded
#pragma unroll
      for (int j = 0; j < 16; ++j)
        BsT[bnb + j][bkr] =
            (_Float16)((kg < 88) ? W1[(size_t)kg * 256 + n0 + bnb + j] : 0.f);
    }
    __syncthreads();

    Frag16 a[2], bfr[2];
#pragma unroll
    for (int mi = 0; mi < 2; ++mi) {
      const _Float16* s = &As[wm + 16*mi + l15][lhi * 8];
      a[mi].f4[0] = *reinterpret_cast<const float4*>(s);
      a[mi].f4[1] = *reinterpret_cast<const float4*>(s + 16);
    }
#pragma unroll
    for (int ni = 0; ni < 2; ++ni) {
      const _Float16* s = &BsT[wn + 16*ni + l15][lhi * 8];
      bfr[ni].f4[0] = *reinterpret_cast<const float4*>(s);
      bfr[ni].f4[1] = *reinterpret_cast<const float4*>(s + 16);
    }
#pragma unroll
    for (int mi = 0; mi < 2; ++mi)
#pragma unroll
      for (int ni = 0; ni < 2; ++ni)
        acc[mi][ni] = __builtin_amdgcn_wmma_f32_16x16x32_f16(
            false, a[mi].v, false, bfr[ni].v, (short)0, acc[mi][ni], false, false);
  }

  // fused relu + mean-pool(4): 4 consecutive pre-pool rows live in one lane
#pragma unroll
  for (int mi = 0; mi < 2; ++mi) {
#pragma unroll
    for (int ni = 0; ni < 2; ++ni) {
      const int   col = n0 + wn + 16*ni + l15;
      const float bv  = B1[col];
#pragma unroll
      for (int r0 = 0; r0 < 8; r0 += 4) {
        float s = 0.f;
#pragma unroll
        for (int r = 0; r < 4; ++r) s += fmaxf(acc[mi][ni][r0 + r] + bv, 0.f);
        const int prow = (m0 + wm + 16*mi + 8*lhi + r0) >> 2;
        Hout[((size_t)b * LP + prow) * DM + col] = 0.25f * s;
      }
    }
  }
}

// ---------------------------------------------------------------------------
// row centering: y = x - mean(x) over 256 cols
// ---------------------------------------------------------------------------
__global__ void __launch_bounds__(256)
k_center(const float* __restrict__ X, float* __restrict__ O)
{
  __shared__ float red[256];
  const size_t row = blockIdx.x;
  const int t = threadIdx.x;
  float v = X[row * 256 + t];
  red[t] = v;
  __syncthreads();
  for (int s = 128; s > 0; s >>= 1) { if (t < s) red[t] += red[t + s]; __syncthreads(); }
  O[row * 256 + t] = v - red[0] * (1.f / 256.f);
}

// ---------------------------------------------------------------------------
// fourier feature map: zp = q[.,h,:32] @ fproj[32,32]; out f16 [cos,sin]*sqrt(2/64)
// ---------------------------------------------------------------------------
__global__ void __launch_bounds__(256)
k_phi(const float* __restrict__ X, const float* __restrict__ FP,
      _Float16* __restrict__ OF)
{
  __shared__ float xs[256];
  const size_t row = blockIdx.x;           // b*LP + l
  const int t = threadIdx.x;
  xs[t] = X[row * 256 + t];
  __syncthreads();
  const int h = t >> 5, mp = t & 31;
  float zp = 0.f;
#pragma unroll
  for (int d = 0; d < 32; ++d) zp += xs[h * 32 + d] * FP[d * 32 + mp];
  const float s = 0.17677669529663687f;    // sqrt(2/64)
  _Float16* o = OF + (row * NH + h) * 64;
  o[mp]      = (_Float16)(cosf(zp) * s);
  o[mp + 32] = (_Float16)(sinf(zp) * s);
}

// positional slope: slope_j/(L-1) = ((256-4j)/128)/2495
__device__ __forceinline__ float pos_ns(int j) {
  return (256.f - 4.f * (float)j) * (1.f / (128.f * 2495.f));
}

// ---------------------------------------------------------------------------
// linear-attention state: kv[b,h,m,d] = sum_l kt[l,m]*v[l,d]
// kt[l,m] = kf[l, m&63] * exp(-hw * ns_j * pos(l,m))   (computed on the fly)
// grid B*NH, 128 threads, tile 128(m) x 32(d), K=L in steps of 32.
// ---------------------------------------------------------------------------
__global__ void __launch_bounds__(128)
k_attn_kv(const _Float16* __restrict__ KF, const float* __restrict__ V,
          const float* __restrict__ HW, float* __restrict__ KVo)
{
  __shared__ __align__(16) _Float16 As[128][40];
  __shared__ __align__(16) _Float16 BsT[32][40];

  const int bh = blockIdx.x;
  const int b  = bh >> 3, h = bh & 7;
  const float hw = HW[h];
  const int tid  = threadIdx.x;
  const int lane = tid & 31;
  const int w    = tid >> 5;
  const int l15  = lane & 15;
  const int lhi  = lane >> 4;

  v8f acc[2][2];
  acc[0][0] = vzero8(); acc[0][1] = vzero8();
  acc[1][0] = vzero8(); acc[1][1] = vzero8();

  const int   m   = tid;                  // A-stage feature row 0..127
  const int   j   = m & 63;
  const float ns  = pos_ns(j);
  const float dir = (m < 64) ? 1.f : -1.f;
  const float gs  = expf(-hw * ns * dir); // geometric step over l
  const int   bl  = tid >> 2;
  const int   bdb = (tid & 3) * 8;

  for (int l0 = 0; l0 < LP; l0 += 32) {
    __syncthreads();
    {  // A: gated kf tile
      float pos0 = (m < 64) ? (float)l0 : (float)(LP - 1 - l0);
      float g    = expf(-hw * ns * pos0);
      const _Float16* kp = KF + (((size_t)b * LP + l0) * NH + h) * 64 + j;
      for (int l = 0; l < 32; ++l) {
        As[m][l] = (_Float16)((float)kp[(size_t)l * (NH * 64)] * g);
        g *= gs;
      }
    }
    {  // B: v tile transposed
      const float* vp = V + ((size_t)b * LP + l0 + bl) * DM + h * 32 + bdb;
#pragma unroll
      for (int jj = 0; jj < 8; ++jj) BsT[bdb + jj][bl] = (_Float16)vp[jj];
    }
    __syncthreads();

    Frag16 a[2], bfr[2];
#pragma unroll
    for (int mi = 0; mi < 2; ++mi) {
      const _Float16* s = &As[32 * w + 16 * mi + l15][lhi * 8];
      a[mi].f4[0] = *reinterpret_cast<const float4*>(s);
      a[mi].f4[1] = *reinterpret_cast<const float4*>(s + 16);
    }
#pragma unroll
    for (int ni = 0; ni < 2; ++ni) {
      const _Float16* s = &BsT[16 * ni + l15][lhi * 8];
      bfr[ni].f4[0] = *reinterpret_cast<const float4*>(s);
      bfr[ni].f4[1] = *reinterpret_cast<const float4*>(s + 16);
    }
#pragma unroll
    for (int mi = 0; mi < 2; ++mi)
#pragma unroll
      for (int ni = 0; ni < 2; ++ni)
        acc[mi][ni] = __builtin_amdgcn_wmma_f32_16x16x32_f16(
            false, a[mi].v, false, bfr[ni].v, (short)0, acc[mi][ni], false, false);
  }

#pragma unroll
  for (int mi = 0; mi < 2; ++mi)
#pragma unroll
    for (int ni = 0; ni < 2; ++ni) {
      const int d = 16 * ni + l15;
#pragma unroll
      for (int r = 0; r < 8; ++r) {
        const int md = 32 * w + 16 * mi + 8 * lhi + r;
        KVo[((size_t)bh * 128 + md) * 32 + d] = acc[mi][ni][r];
      }
    }
}

// ---------------------------------------------------------------------------
// readout: y[b,l,h*32+d] = (1/L) sum_m qt[l,m]*kv[m,d]
// qt[l,m] = qf[l, m&63] * exp(+hw * ns_j * pos(l,m))
// grid (B*NH, LP/64), 64 threads (2 waves), tile 64(l) x 32(d), K=128.
// ---------------------------------------------------------------------------
__global__ void __launch_bounds__(64)
k_attn_out(const _Float16* __restrict__ QF, const float* __restrict__ KVi,
           const float* __restrict__ HW, float* __restrict__ Y)
{
  __shared__ __align__(16) _Float16 As[64][40];
  __shared__ __align__(16) _Float16 BsT[32][40];

  const int bh = blockIdx.x;
  const int b  = bh >> 3, h = bh & 7;
  const int l0 = blockIdx.y * 64;
  const float hw = HW[h];
  const int tid  = threadIdx.x;
  const int lane = tid & 31;
  const int w    = tid >> 5;
  const int l15  = lane & 15;
  const int lhi  = lane >> 4;

  v8f acc[2][2];
  acc[0][0] = vzero8(); acc[0][1] = vzero8();
  acc[1][0] = vzero8(); acc[1][1] = vzero8();

  const int al = tid;                      // A-stage sequence row
  const int lg = l0 + al;
  const int bkr = tid >> 1;
  const int bdb = (tid & 1) * 16;

  for (int kk = 0; kk < 128; kk += 32) {
    __syncthreads();
    {  // A: gated qf tile
      const _Float16* qp = QF + (((size_t)b * LP + lg) * NH + h) * 64;
#pragma unroll
      for (int kl = 0; kl < 32; ++kl) {
        int   mm  = kk + kl;
        int   jj  = mm & 63;
        float pos = (mm < 64) ? (float)lg : (float)(LP - 1 - lg);
        As[al][kl] = (_Float16)((float)qp[jj] * expf(hw * pos_ns(jj) * pos));
      }
    }
    {  // B: kv tile transposed
      const float* kvp = KVi + ((size_t)bh * 128 + kk + bkr) * 32 + bdb;
#pragma unroll
      for (int jj = 0; jj < 16; ++jj) BsT[bdb + jj][bkr] = (_Float16)kvp[jj];
    }
    __syncthreads();

    Frag16 a[2], bfr[2];
#pragma unroll
    for (int mi = 0; mi < 2; ++mi) {
      const _Float16* s = &As[32 * w + 16 * mi + l15][lhi * 8];
      a[mi].f4[0] = *reinterpret_cast<const float4*>(s);
      a[mi].f4[1] = *reinterpret_cast<const float4*>(s + 16);
    }
#pragma unroll
    for (int ni = 0; ni < 2; ++ni) {
      const _Float16* s = &BsT[16 * ni + l15][lhi * 8];
      bfr[ni].f4[0] = *reinterpret_cast<const float4*>(s);
      bfr[ni].f4[1] = *reinterpret_cast<const float4*>(s + 16);
    }
#pragma unroll
    for (int mi = 0; mi < 2; ++mi)
#pragma unroll
      for (int ni = 0; ni < 2; ++ni)
        acc[mi][ni] = __builtin_amdgcn_wmma_f32_16x16x32_f16(
            false, a[mi].v, false, bfr[ni].v, (short)0, acc[mi][ni], false, false);
  }

  const float invL = 1.f / (float)LP;
#pragma unroll
  for (int mi = 0; mi < 2; ++mi)
#pragma unroll
    for (int ni = 0; ni < 2; ++ni) {
      const int col = h * 32 + 16 * ni + l15;
#pragma unroll
      for (int r = 0; r < 8; ++r) {
        const int lr = l0 + 32 * w + 16 * mi + 8 * lhi + r;
        Y[((size_t)b * LP + lr) * DM + col] = acc[mi][ni][r] * invL;
      }
    }
}

// ---------------------------------------------------------------------------
// softmax pooling pieces
// ---------------------------------------------------------------------------
__global__ void __launch_bounds__(256)
k_score(const float* __restrict__ KH, const float* __restrict__ SQ,
        float* __restrict__ SCO)
{
  __shared__ float red[256];
  const size_t row = blockIdx.x;           // b*LP + n
  const int t = threadIdx.x;
  red[t] = KH[row * 256 + t] * SQ[t];      // col t = d*8+h matches sm_q flat
  __syncthreads();
  if (t < 8) {
    float s = 0.f;
    for (int d = 0; d < 32; ++d) s += red[d * 8 + t];
    SCO[row * 8 + t] = s * 0.17677669529663687f;   // 32^-0.5 * smoothing
  }
}

__global__ void __launch_bounds__(256)
k_softmax(float* __restrict__ SCO)
{
  __shared__ float red[256];
  const int b = blockIdx.x >> 3, h = blockIdx.x & 7;
  float* p = SCO + (size_t)b * LP * 8 + h;
  const int t = threadIdx.x;
  float mx = -1e30f;
  for (int n = t; n < LP; n += 256) mx = fmaxf(mx, p[(size_t)n * 8]);
  red[t] = mx; __syncthreads();
  for (int s = 128; s > 0; s >>= 1) { if (t < s) red[t] = fmaxf(red[t], red[t + s]); __syncthreads(); }
  mx = red[0]; __syncthreads();
  float sm = 0.f;
  for (int n = t; n < LP; n += 256) sm += expf(p[(size_t)n * 8] - mx);
  red[t] = sm; __syncthreads();
  for (int s = 128; s > 0; s >>= 1) { if (t < s) red[t] += red[t + s]; __syncthreads(); }
  const float inv = 1.f / red[0];
  for (int n = t; n < LP; n += 256) p[(size_t)n * 8] = expf(p[(size_t)n * 8] - mx) * inv;
}

__global__ void __launch_bounds__(256)
k_pool(const float* __restrict__ VH, const float* __restrict__ PR,
       float* __restrict__ PO)
{
  __shared__ float red[256];
  const int b  = blockIdx.x >> 8;
  const int oc = blockIdx.x & 255;         // oc = h*32 + d
  const int d  = oc & 31, h = oc >> 5;
  const int col = d * 8 + h;               // vh flat col
  const int t = threadIdx.x;
  float s = 0.f;
  for (int n = t; n < LP; n += 256)
    s += VH[((size_t)b * LP + n) * 256 + col] * PR[((size_t)b * LP + n) * 8 + h];
  red[t] = s; __syncthreads();
  for (int st = 128; st > 0; st >>= 1) { if (t < st) red[t] += red[t + st]; __syncthreads(); }
  if (t == 0) PO[b * 256 + oc] = red[0];
}

// ---------------------------------------------------------------------------
// host launcher
// ---------------------------------------------------------------------------
extern "C" void kernel_launch(void* const* d_in, const int* in_sizes, int n_in,
                              void* d_out, int out_size, void* d_ws, size_t ws_size,
                              hipStream_t stream)
{
  (void)in_sizes; (void)n_in; (void)out_size; (void)ws_size;

  const float* x   = (const float*)d_in[0];
  const float* c0w = (const float*)d_in[1];
  const float* c0b = (const float*)d_in[2];
  const float* c1w = (const float*)d_in[3];
  const float* c1b = (const float*)d_in[4];
  const float* smq  = (const float*)d_in[33];
  const float* smkw = (const float*)d_in[34];
  const float* smkb = (const float*)d_in[35];
  const float* smvw = (const float*)d_in[36];
  const float* smvb = (const float*)d_in[37];
  const float* fcw  = (const float*)d_in[38];
  const float* fcb  = (const float*)d_in[39];

  float* ws = (float*)d_ws;
  // workspace layout (floats); total ~86.5M floats (~346 MB)
  const size_t SZ_H0 = (size_t)BB * P0 * C0;     // 1,279,616
  const size_t SZ_BL = (size_t)M2 * DM;          // 10,223,616
  const size_t SZ_KV = (size_t)BB * NH * 128 * 32;
  const size_t SZ_T1 = (size_t)LP * 1024;
  const size_t SZ_SC = (size_t)M2 * NH;

  float* H0b = ws;
  float* HB  = H0b + SZ_H0;
  float* XB  = HB  + SZ_BL;
  float* QB  = XB  + SZ_BL;
  float* KB  = QB  + SZ_BL;
  float* VB  = KB  + SZ_BL;
  float* YB  = VB  + SZ_BL;
  _Float16* QF = (_Float16*)(YB + SZ_BL);        // M2*NH*64 halves
  _Float16* KF = QF + (size_t)M2 * NH * 64;
  float* KV  = (float*)(KF + (size_t)M2 * NH * 64);
  float* T1  = KV + SZ_KV;
  float* SC  = T1 + SZ_T1;
  float* PL  = SC + SZ_SC;

  // ---- conv frontend ----
  k_conv0<<<(BB * P0 * C0 + 255) / 256, 256, 0, stream>>>(x, c0w, c0b, H0b);
  k_conv1<<<dim3(4, 156, BB), 128, 0, stream>>>(H0b, c1w, c1b, HB);

  // ---- transformer layers ----
  for (int i = 0; i < 2; ++i) {
    const int base = 5 + i * 14;
    const float* qw  = (const float*)d_in[base + 0];
    const float* qb  = (const float*)d_in[base + 1];
    const float* kw  = (const float*)d_in[base + 2];
    const float* kb  = (const float*)d_in[base + 3];
    const float* vw  = (const float*)d_in[base + 4];
    const float* vb  = (const float*)d_in[base + 5];
    const float* ow  = (const float*)d_in[base + 6];
    const float* ob  = (const float*)d_in[base + 7];
    const float* fp  = (const float*)d_in[base + 8];
    const float* hwt = (const float*)d_in[base + 9];
    const float* f1w = (const float*)d_in[base + 10];
    const float* f1b = (const float*)d_in[base + 11];
    const float* f2w = (const float*)d_in[base + 12];
    const float* f2b = (const float*)d_in[base + 13];

    k_center<<<M2, 256, 0, stream>>>(HB, XB);
    k_gemm64<<<dim3(4, M2 / 64), 128, 0, stream>>>(XB, DM, qw, qb, nullptr, QB, DM, M2, DM, DM, 0);
    k_gemm64<<<dim3(4, M2 / 64), 128, 0, stream>>>(XB, DM, kw, kb, nullptr, KB, DM, M2, DM, DM, 0);
    k_gemm64<<<dim3(4, M2 / 64), 128, 0, stream>>>(XB, DM, vw, vb, nullptr, VB, DM, M2, DM, DM, 0);
    k_phi<<<M2, 256, 0, stream>>>(QB, fp, QF);
    k_phi<<<M2, 256, 0, stream>>>(KB, fp, KF);
    k_attn_kv<<<BB * NH, 128, 0, stream>>>(KF, VB, hwt, KV);
    k_attn_out<<<dim3(BB * NH, LP / 64), 64, 0, stream>>>(QF, KV, hwt, YB);
    // a = oproj(y) + h  -> XB
    k_gemm64<<<dim3(4, M2 / 64), 128, 0, stream>>>(YB, DM, ow, ob, HB, XB, DM, M2, DM, DM, 0);
    // ff: center(a) -> QB; per-batch GEMM chunks through T1
    k_center<<<M2, 256, 0, stream>>>(XB, QB);
    for (int b = 0; b < BB; ++b) {
      const size_t off = (size_t)b * LP * DM;
      k_gemm64<<<dim3(16, LP / 64), 128, 0, stream>>>(QB + off, DM, f1w, f1b, nullptr,
                                                      T1, 1024, LP, 1024, DM, 1);
      k_gemm64<<<dim3(4, LP / 64), 128, 0, stream>>>(T1, 1024, f2w, f2b, XB + off,
                                                     HB + off, DM, LP, DM, 1024, 0);
    }
  }

  // ---- softmax-attention pooling + classifier ----
  k_gemm64<<<dim3(4, M2 / 64), 128, 0, stream>>>(HB, DM, smkw, smkb, nullptr, QB, DM, M2, DM, DM, 0);
  k_gemm64<<<dim3(4, M2 / 64), 128, 0, stream>>>(HB, DM, smvw, smvb, nullptr, KB, DM, M2, DM, DM, 0);
  k_score<<<M2, 256, 0, stream>>>(QB, smq, SC);
  k_softmax<<<BB * NH, 256, 0, stream>>>(SC);
  k_pool<<<BB * 256, 256, 0, stream>>>(KB, SC, PL);
  k_gemm64<<<dim3(4, 1), 128, 0, stream>>>(PL, DM, fcw, fcb, nullptr, (float*)d_out, DM,
                                           BB, DM, DM, 0);
}
